// ClauseExtractor_42099269435626
// MI455X (gfx1250) — compile-verified, matrix operand order checked by reading, not run on hardware
//
#include <hip/hip_runtime.h>

typedef __attribute__((ext_vector_type(16))) _Float16 v16h;
typedef __attribute__((ext_vector_type(8)))  float    v8f;

#define B_ 8
#define S_ 512
#define D_ 256
#define MAX_SPAN_ 100
#define NUM_TYPES_ 15
#define NSPAN 46250          // total spans per batch for S=512, MAX_SPAN=100

#define MT 64                // rows (spans) per workgroup
#define NMT (MT / 16)        // 4 WMMA M-tiles

// LDS row strides (halves), padded to dodge bank conflicts
#define XS 528               // 512 + 16
#define HS 264               // 256 + 8
#define PS 136               // 128 + 8

// ---------------------------------------------------------------------------
// Prep kernel 1: span index tables (starts[n], ends[n]) for n in [0, NSPAN)
// ---------------------------------------------------------------------------
__global__ void build_indices(int* __restrict__ starts, int* __restrict__ ends) {
  int i = blockIdx.x * blockDim.x + threadIdx.x;
  if (i >= S_) return;
  int cnt = (S_ - i < MAX_SPAN_) ? (S_ - i) : MAX_SPAN_;
  const int F = S_ - MAX_SPAN_ + 1;          // starts with a full MAX_SPAN run (=413)
  long long off;
  if (i < F) {
    off = (long long)i * MAX_SPAN_;
  } else {
    long long tri = ((long long)(MAX_SPAN_ - 1) * MAX_SPAN_
                   - (long long)(S_ - i) * (S_ - i + 1)) / 2;
    off = (long long)F * MAX_SPAN_ + tri;
  }
  for (int k = 0; k < cnt; ++k) {
    starts[off + k] = i;
    ends[off + k]   = i + k;
  }
}

// ---------------------------------------------------------------------------
// Prep kernel 2: transpose + f32->f16 weights into K-contiguous (N x K) layout
// ---------------------------------------------------------------------------
__global__ void convert_weights(const float* __restrict__ W_enc,
                                const float* __restrict__ W_s1,
                                const float* __restrict__ W_c1,
                                _Float16* __restrict__ WencT,
                                _Float16* __restrict__ Ws1T,
                                _Float16* __restrict__ Wc1T) {
  int idx = blockIdx.x * blockDim.x + threadIdx.x;
  const int NE = 2 * D_ * D_;        // 131072
  const int NS = D_ * (D_ / 2);      // 32768
  if (idx < NE) {
    int n = idx >> 9;                // / 512
    int k = idx & 511;
    WencT[idx] = (_Float16)W_enc[k * D_ + n];
  } else if (idx < NE + NS) {
    int j = idx - NE;
    int n = j >> 8;                  // / 256  (n < 128)
    int k = j & 255;
    Ws1T[j] = (_Float16)W_s1[k * (D_ / 2) + n];
    Wc1T[j] = (_Float16)W_c1[k * (D_ / 2) + n];
  }
}

// ---------------------------------------------------------------------------
// Fragment loader for V_WMMA_F32_16X16X32_F16 A/B operands.
// Row-major [r][k] source, K contiguous (r = M for A, r = N for B^T).
// 16-bit A-matrix 16x32 lane layout (wave32):
//   lanes 0-15 : r=lane,    halves 0..7  -> K=kb+0..7,  halves 8..15 -> K=kb+16..23
//   lanes 16-31: r=lane-16, halves 0..7  -> K=kb+8..15, halves 8..15 -> K=kb+24..31
// ---------------------------------------------------------------------------
__device__ inline v16h frag_ld(const _Float16* __restrict__ p, int stride,
                               int kb, int lane) {
  int r  = lane & 15;
  int kh = (lane >> 4) << 3;         // 0 or 8
  const _Float16* q = p + r * stride + kb + kh;
  v16h f;
#pragma unroll
  for (int v = 0; v < 4; ++v) { f[2 * v] = q[2 * v]; f[2 * v + 1] = q[2 * v + 1]; }
  q += 16;
#pragma unroll
  for (int v = 0; v < 4; ++v) { f[8 + 2 * v] = q[2 * v]; f[9 + 2 * v] = q[2 * v + 1]; }
  return f;
}

__device__ inline v8f wmma16(v16h a, v16h b, v8f c) {
  return __builtin_amdgcn_wmma_f32_16x16x32_f16(false, a, false, b,
                                                (short)0, c, false, false);
}

// ---------------------------------------------------------------------------
// Main kernel: 64-span tile per workgroup, 256 threads = 8 waves.
//   Stage 1: h = relu(X*W_enc + b_enc) + width_table   (K=512, N=256)
//   Stage 2: s_hid / c_hid = relu(h*W)                 (K=256, N=128 each)
//   Stage 3: tiny heads (dot-128) via VALU
// Each wave keeps 4 M-tile accumulators so every B fragment is reused 4x.
// ---------------------------------------------------------------------------
__global__ __launch_bounds__(256) void span_kernel(
    const float* __restrict__ emb, const _Float16* __restrict__ WencT,
    const float* __restrict__ b_enc, const float* __restrict__ wtab,
    const _Float16* __restrict__ Ws1T, const float* __restrict__ b_s1,
    const float* __restrict__ W_s2, const float* __restrict__ b_s2,
    const _Float16* __restrict__ Wc1T, const float* __restrict__ b_c1,
    const float* __restrict__ W_c2, const float* __restrict__ b_c2,
    const int* __restrict__ starts, const int* __restrict__ ends,
    float* __restrict__ out_scores, float* __restrict__ out_logits) {
  __shared__ _Float16 Xl[MT * XS];   // stage 1 input tile; reused for s/c in stage 2+
  __shared__ _Float16 Hl[MT * HS];   // h (f16), MT x 256
  __shared__ int st_sh[MT], en_sh[MT], wd_sh[MT];

  _Float16* Sl = Xl;                 // MT x PS  (aliases dead X buffer)
  _Float16* Cl = Xl + MT * PS;       // MT x PS

  const int tid  = threadIdx.x;
  const int lane = tid & 31;
  const int wv   = tid >> 5;
  const int b    = blockIdx.y;
  const int row0 = blockIdx.x * MT;

  if (tid < MT) {
    int r  = row0 + tid;
    int rc = (r < NSPAN) ? r : (NSPAN - 1);   // clamp for partial tile
    int s = starts[rc], e = ends[rc];
    st_sh[tid] = s; en_sh[tid] = e; wd_sh[tid] = e - s;
  }
  __syncthreads();

  // ---- Gather X tile (f32 -> f16): cols 0..255 = emb[start], 256..511 = emb[end]
  const float4* emb4 = (const float4*)(emb + (size_t)b * S_ * D_);
  for (int idx = tid; idx < MT * 128; idx += 256) {
    int m  = idx >> 7;               // row
    int c4 = idx & 127;              // float4 column
    int tok = (c4 < 64) ? st_sh[m] : en_sh[m];
    float4 v = emb4[(size_t)tok * 64 + (c4 & 63)];
    _Float16* dst = &Xl[m * XS + c4 * 4];
    dst[0] = (_Float16)v.x; dst[1] = (_Float16)v.y;
    dst[2] = (_Float16)v.z; dst[3] = (_Float16)v.w;
  }
  __syncthreads();

  // ---- Stage 1: wave wv owns h columns [32*wv, 32*wv+32), 4 M-tiles deep
  {
    const int nb0 = wv * 32, nb1 = wv * 32 + 16;
    const _Float16* B0 = WencT + (size_t)nb0 * 512;
    const _Float16* B1 = WencT + (size_t)nb1 * 512;
    __builtin_prefetch(B0, 0, 3);
    __builtin_prefetch(B1, 0, 3);
    v8f acc0[NMT] = {}; v8f acc1[NMT] = {};
#pragma unroll 2
    for (int kb = 0; kb < 512; kb += 32) {
      v16h w0 = frag_ld(B0, 512, kb, lane);
      v16h w1 = frag_ld(B1, 512, kb, lane);
#pragma unroll
      for (int mt = 0; mt < NMT; ++mt) {
        v16h a = frag_ld(Xl + mt * 16 * XS, XS, kb, lane);
        acc0[mt] = wmma16(a, w0, acc0[mt]);
        acc1[mt] = wmma16(a, w1, acc1[mt]);
      }
    }
    const int nl   = lane & 15;
    const int mh   = (lane >> 4) << 3;
    const int col0 = nb0 + nl, col1 = nb1 + nl;
    const float be0 = b_enc[col0], be1 = b_enc[col1];
#pragma unroll
    for (int mt = 0; mt < NMT; ++mt) {
#pragma unroll
      for (int v = 0; v < 8; ++v) {
        int m = mt * 16 + v + mh;
        int w = wd_sh[m];
        float h0 = fmaxf(acc0[mt][v] + be0, 0.f) + wtab[w * D_ + col0];
        float h1 = fmaxf(acc1[mt][v] + be1, 0.f) + wtab[w * D_ + col1];
        Hl[m * HS + col0] = (_Float16)h0;
        Hl[m * HS + col1] = (_Float16)h1;
      }
    }
  }
  __syncthreads();   // Hl ready; Xl dead from here (Sl/Cl may overwrite)

  // ---- Stage 2: wave wv owns columns [16*wv, 16*wv+16) of both heads
  {
    const int nb = wv * 16;
    const _Float16* Bs = Ws1T + (size_t)nb * 256;
    const _Float16* Bc = Wc1T + (size_t)nb * 256;
    __builtin_prefetch(Bs, 0, 3);
    __builtin_prefetch(Bc, 0, 3);
    v8f cs[NMT] = {}; v8f cc[NMT] = {};
#pragma unroll 2
    for (int kb = 0; kb < 256; kb += 32) {
      v16h ws = frag_ld(Bs, 256, kb, lane);
      v16h wc = frag_ld(Bc, 256, kb, lane);
#pragma unroll
      for (int mt = 0; mt < NMT; ++mt) {
        v16h a = frag_ld(Hl + mt * 16 * HS, HS, kb, lane);
        cs[mt] = wmma16(a, ws, cs[mt]);
        cc[mt] = wmma16(a, wc, cc[mt]);
      }
    }
    const int nl  = lane & 15;
    const int mh  = (lane >> 4) << 3;
    const int col = nb + nl;
    const float bs1 = b_s1[col], bc1 = b_c1[col];
#pragma unroll
    for (int mt = 0; mt < NMT; ++mt) {
#pragma unroll
      for (int v = 0; v < 8; ++v) {
        int m = mt * 16 + v + mh;
        Sl[m * PS + col] = (_Float16)fmaxf(cs[mt][v] + bs1, 0.f);
        Cl[m * PS + col] = (_Float16)fmaxf(cc[mt][v] + bc1, 0.f);
      }
    }
  }
  __syncthreads();

  // ---- Stage 3: tiny heads via VALU dots from LDS
  if (tid < MT) {                          // span score: MT rows x dot(128)
    int m = tid, row = row0 + m;
    if (row < NSPAN) {
      float acc = b_s2[0];
#pragma unroll 8
      for (int k = 0; k < D_ / 2; ++k) acc += (float)Sl[m * PS + k] * W_s2[k];
      out_scores[(size_t)b * NSPAN + row] = acc;
    }
  }
  for (int o = tid; o < MT * NUM_TYPES_; o += 256) {   // logits: 960 outputs
    int m = o / NUM_TYPES_, t = o % NUM_TYPES_;
    int row = row0 + m;
    if (row < NSPAN) {
      float acc = b_c2[t];
#pragma unroll 8
      for (int k = 0; k < D_ / 2; ++k)
        acc += (float)Cl[m * PS + k] * W_c2[k * NUM_TYPES_ + t];
      out_logits[((size_t)b * NSPAN + row) * NUM_TYPES_ + t] = acc;
    }
  }
}

// ---------------------------------------------------------------------------
extern "C" void kernel_launch(void* const* d_in, const int* in_sizes, int n_in,
                              void* d_out, int out_size, void* d_ws, size_t ws_size,
                              hipStream_t stream) {
  const float* emb   = (const float*)d_in[0];
  const float* W_enc = (const float*)d_in[1];
  const float* b_enc = (const float*)d_in[2];
  const float* wtab  = (const float*)d_in[3];
  const float* W_s1  = (const float*)d_in[4];
  const float* b_s1  = (const float*)d_in[5];
  const float* W_s2  = (const float*)d_in[6];
  const float* b_s2  = (const float*)d_in[7];
  const float* W_c1  = (const float*)d_in[8];
  const float* b_c1  = (const float*)d_in[9];
  const float* W_c2  = (const float*)d_in[10];
  const float* b_c2  = (const float*)d_in[11];

  // workspace layout (bytes, 256-aligned): total ~746 KB
  char* ws = (char*)d_ws;
  int*      starts = (int*)(ws + 0);           // 46250 * 4 = 185000
  int*      ends   = (int*)(ws + 185088);      // 185000
  _Float16* WencT  = (_Float16*)(ws + 370176); // 256*512*2 = 262144
  _Float16* Ws1T   = (_Float16*)(ws + 632320); // 128*256*2 = 65536
  _Float16* Wc1T   = (_Float16*)(ws + 697856); // 65536

  build_indices<<<2, 256, 0, stream>>>(starts, ends);

  const int convN = 2 * D_ * D_ + D_ * (D_ / 2);
  convert_weights<<<(convN + 255) / 256, 256, 0, stream>>>(W_enc, W_s1, W_c1,
                                                           WencT, Ws1T, Wc1T);

  float* out_scores = (float*)d_out;
  float* out_logits = out_scores + (size_t)B_ * NSPAN;
  dim3 grid((NSPAN + MT - 1) / MT, B_);
  span_kernel<<<grid, 256, 0, stream>>>(emb, WencT, b_enc, wtab,
                                        Ws1T, b_s1, W_s2, b_s2,
                                        Wc1T, b_c1, W_c2, b_c2,
                                        starts, ends, out_scores, out_logits);
}
